// SimpleGCNRec_43396349558973
// MI455X (gfx1250) — matrix operand order repaired and checked by reading, not compiled, not run on hardware
//
#include <hip/hip_runtime.h>
#include <math.h>

typedef __attribute__((ext_vector_type(16))) _Float16 v16h;
typedef __attribute__((ext_vector_type(8)))  _Float16 v8h;
typedef __attribute__((ext_vector_type(4)))  _Float16 v4h;
typedef __attribute__((ext_vector_type(8)))  float    v8f;

#define HID 128
#define SLOPE 0.01f
#define EPS 1e-5f

__device__ __forceinline__ float wave_sum32(float v) {
#pragma unroll
  for (int off = 16; off > 0; off >>= 1)
    v += __shfl_xor(v, off, 32);
  return v;
}

// ---------------- utility kernels ----------------

__global__ void k_zero(float* __restrict__ p, long long n) {
  long long i = (long long)blockIdx.x * blockDim.x + threadIdx.x;
  long long stride = (long long)gridDim.x * blockDim.x;
  for (; i < n; i += stride) p[i] = 0.0f;
}

// deg[dst] += 1 for each edge (self-loop +1 added in k_dinv)
__global__ void k_deg(const int* __restrict__ dst, long long E, float* __restrict__ deg) {
  long long i = (long long)blockIdx.x * blockDim.x + threadIdx.x;
  long long stride = (long long)gridDim.x * blockDim.x;
  for (; i < E; i += stride) atomicAdd(&deg[dst[i]], 1.0f);
}

__global__ void k_dinv(float* __restrict__ deg, int n) {
  int i = blockIdx.x * blockDim.x + threadIdx.x;
  if (i < n) deg[i] = rsqrtf(deg[i] + 1.0f);   // deg includes self loop, always > 0
}

// ---------------- weight pre-pack into WMMA B-fragment order ----------------
// Bp[((tile*(K/32) + kk)*32 + lane)*16 + e] = (f16) W[(kk*32 + 16*(lane>>4) + e)*128 + tile*16 + (lane&15)]
// so the GEMM loads each lane's 16 halves as one aligned v16h (2 x b128).

__global__ void k_packB(const float* __restrict__ W, _Float16* __restrict__ Bp, int K) {
  const int kk_n = K >> 5;              // k-steps of 32
  const int total = 128 * K;            // 8 tiles * kk_n * 32 lanes * 16 elems
  int i = blockIdx.x * blockDim.x + threadIdx.x;
  int stride = gridDim.x * blockDim.x;
  for (; i < total; i += stride) {
    int e    = i & 15;
    int lane = (i >> 4) & 31;
    int grp  = i >> 9;                  // tile*kk_n + kk
    int kk   = grp % kk_n;
    int tile = grp / kk_n;
    int k = kk * 32 + 16 * (lane >> 4) + e;
    int n = tile * 16 + (lane & 15);
    Bp[i] = (_Float16)W[k * HID + n];
  }
}

// ---------------- WMMA GEMM: Y[M x 128] = A[M x K] * W[K x 128] (+bias) ----------------
// One 16x16 output tile per wave, 8 waves/block cover the 128-wide output row-block.
// A tile (16 x K) staged in LDS as f16 once per block; B pre-packed in fragment order.

template <bool ADD_BIAS, int K>
__global__ void k_gemm_wmma(const float* __restrict__ A, const _Float16* __restrict__ Bp,
                            const float* __restrict__ bias, float* __restrict__ Y, int M) {
  __shared__ _Float16 As[16 * K];

  const int lane = threadIdx.x & 31;
  const int wave = threadIdx.x >> 5;
  const int l16  = lane & 15;
  const int half = lane >> 4;
  const int m0   = blockIdx.x * 16;
  const int n0   = wave * 16;
  constexpr int KK = K / 32;            // k-steps
  constexpr int KV = K / 4;             // float4 per row

  // cooperative stage of the 16 x K A-tile (fp32 -> f16), vectorized
  for (int i = threadIdx.x; i < 16 * KV; i += blockDim.x) {
    int r = i / KV, c = i % KV;         // compile-time power-of-two -> shifts
    float4 v = *(const float4*)(A + (long long)(m0 + r) * K + c * 4);
    v4h h;
    h[0] = (_Float16)v.x; h[1] = (_Float16)v.y;
    h[2] = (_Float16)v.z; h[3] = (_Float16)v.w;
    *(v4h*)(&As[r * K + c * 4]) = h;
  }
  __syncthreads();

  v8f acc = {};
  const v16h* bp = (const v16h*)(Bp) + (wave * KK) * 32 + lane;
#pragma unroll
  for (int kk = 0; kk < KK; ++kk) {
    const int k0 = kk * 32;
    // A fragment: row M=l16; elems 0..15 -> K=k0+16*half+e
    const v8h alo = *(const v8h*)(&As[l16 * K + k0 + 16 * half]);
    const v8h ahi = *(const v8h*)(&As[l16 * K + k0 + 16 * half + 8]);
    v16h a;
#pragma unroll
    for (int e = 0; e < 8; ++e) { a[e] = alo[e]; a[e + 8] = ahi[e]; }
    // B fragment: one aligned 32B load from the packed layout
    const v16h b = bp[kk * 32];

    acc = __builtin_amdgcn_wmma_f32_16x16x32_f16(
        /*neg_a=*/false, a, /*neg_b=*/false, b,
        /*c_mod=*/(short)0, acc, /*reuse_a=*/false, /*reuse_b=*/false);
  }

  const float bv = ADD_BIAS ? bias[n0 + l16] : 0.0f;
  // C/D layout: VGPR r -> row m0 + r + 8*half, col n0 + l16
#pragma unroll
  for (int r = 0; r < 8; ++r) {
    int row = m0 + r + 8 * half;
    Y[(long long)row * HID + n0 + l16] = acc[r] + bv;
  }
}

// ---------------- edge aggregation: agg[dst] += xw[src] * dinv[src]*dinv[dst] ----------------
// one wave per edge; each lane owns 4 of the 128 dims (b128 gather + 4 f32 atomics)

__global__ void k_aggregate(const float* __restrict__ xw, const float* __restrict__ dinv,
                            const int* __restrict__ src, const int* __restrict__ dst,
                            float* __restrict__ agg, long long E) {
  const int lane = threadIdx.x & 31;
  long long wid = ((long long)blockIdx.x * blockDim.x + threadIdx.x) >> 5;
  long long nw  = ((long long)gridDim.x * blockDim.x) >> 5;
  for (long long e = wid; e < E; e += nw) {
    int s = src[e], d = dst[e];
    float nrm = dinv[s] * dinv[d];
    const float4 v = *(const float4*)(xw + (long long)s * HID + lane * 4);
    float* out = agg + (long long)d * HID + lane * 4;
    atomicAdd(out + 0, v.x * nrm);
    atomicAdd(out + 1, v.y * nrm);
    atomicAdd(out + 2, v.z * nrm);
    atomicAdd(out + 3, v.w * nrm);
  }
}

// ---------------- self-loop + bias + LayerNorm + leaky ReLU (in place on agg) ----------------

__global__ void k_post(float* __restrict__ x, const float* __restrict__ xw,
                       const float* __restrict__ dinv, const float* __restrict__ bias,
                       const float* __restrict__ g, const float* __restrict__ beta, int N) {
  const int lane = threadIdx.x & 31;
  const int d0 = lane * 4;
  long long wid = ((long long)blockIdx.x * blockDim.x + threadIdx.x) >> 5;
  long long nw  = ((long long)gridDim.x * blockDim.x) >> 5;
  for (long long n = wid; n < N; n += nw) {
    float di = dinv[n];
    float self = di * di;
    float v[4];
    float s = 0.0f, ssq = 0.0f;
#pragma unroll
    for (int j = 0; j < 4; ++j) {
      float t = x[n * HID + d0 + j] + xw[n * HID + d0 + j] * self + bias[d0 + j];
      v[j] = t;
      s += t;
      ssq += t * t;
    }
    s   = wave_sum32(s);
    ssq = wave_sum32(ssq);
    const float mu  = s * (1.0f / HID);
    const float var = ssq * (1.0f / HID) - mu * mu;
    const float rstd = rsqrtf(var + EPS);
#pragma unroll
    for (int j = 0; j < 4; ++j) {
      float y = (v[j] - mu) * rstd * g[d0 + j] + beta[d0 + j];
      x[n * HID + d0 + j] = (y > 0.0f) ? y : SLOPE * y;
    }
  }
}

// ---------------- prediction head: one wave per pair ----------------

__device__ __forceinline__ float sigmoidf(float z) { return 1.0f / (1.0f + expf(-z)); }

__global__ void k_predict(const float* __restrict__ x, const int* __restrict__ eu,
                          const int* __restrict__ ef, const float* __restrict__ eW,
                          const float* __restrict__ eB, const float* __restrict__ rW,
                          const float* __restrict__ rB, float* __restrict__ exist,
                          float* __restrict__ rating, long long P) {
  const int lane = threadIdx.x & 31;
  const int d0 = lane * 4;
  const float4 we0 = *(const float4*)(eW + d0);
  const float4 we1 = *(const float4*)(eW + HID + d0);
  const float4 wr0 = *(const float4*)(rW + d0);
  const float4 wr1 = *(const float4*)(rW + HID + d0);
  const float eb = eB[0], rb = rB[0];

  long long wid = ((long long)blockIdx.x * blockDim.x + threadIdx.x) >> 5;
  long long nw  = ((long long)gridDim.x * blockDim.x) >> 5;
  for (long long p = wid; p < P; p += nw) {
    int u = eu[p], f = ef[p];
    const float4 xu = *(const float4*)(x + (long long)u * HID + d0);
    const float4 xf = *(const float4*)(x + (long long)f * HID + d0);
    float ae = xu.x * we0.x + xu.y * we0.y + xu.z * we0.z + xu.w * we0.w
             + xf.x * we1.x + xf.y * we1.y + xf.z * we1.z + xf.w * we1.w;
    float ar = xu.x * wr0.x + xu.y * wr0.y + xu.z * wr0.z + xu.w * wr0.w
             + xf.x * wr1.x + xf.y * wr1.y + xf.z * wr1.z + xf.w * wr1.w;
    ae = wave_sum32(ae);
    ar = wave_sum32(ar);
    if (lane == 0) {
      exist[p]  = sigmoidf(ae + eb);
      rating[p] = 1.0f + 4.0f * sigmoidf(ar + rb);
    }
  }
}

// ---------------- launcher ----------------

extern "C" void kernel_launch(void* const* d_in, const int* in_sizes, int n_in,
                              void* d_out, int out_size, void* d_ws, size_t ws_size,
                              hipStream_t stream) {
  (void)n_in; (void)out_size; (void)ws_size;

  const float* u_feat = (const float*)d_in[0];
  const float* f_feat = (const float*)d_in[1];
  const int*   eidx   = (const int*)d_in[2];
  const int*   edge_u = (const int*)d_in[3];
  const int*   edge_f = (const int*)d_in[4];
  const float* u_W  = (const float*)d_in[5];
  const float* u_b  = (const float*)d_in[6];
  const float* f_W  = (const float*)d_in[7];
  const float* f_b  = (const float*)d_in[8];
  const float* c1_W = (const float*)d_in[9];
  const float* c1_b = (const float*)d_in[10];
  const float* c2_W = (const float*)d_in[11];
  const float* c2_b = (const float*)d_in[12];
  const float* n1_g = (const float*)d_in[13];
  const float* n1_b = (const float*)d_in[14];
  const float* n2_g = (const float*)d_in[15];
  const float* n2_b = (const float*)d_in[16];
  const float* e_W  = (const float*)d_in[17];
  const float* e_b  = (const float*)d_in[18];
  const float* r_W  = (const float*)d_in[19];
  const float* r_b  = (const float*)d_in[20];

  const int N_U = 50000, N_F = 20000, N = N_U + N_F;
  const long long E = (long long)in_sizes[2] / 2;
  const long long P = (long long)in_sizes[3];
  const int* src = eidx;
  const int* dst = eidx + E;

  // workspace: xw[N*128] f32 | agg[N*128] f32 | dinv[N] f32 | packed f16 weights
  float* xw   = (float*)d_ws;
  float* agg  = xw + (size_t)N * HID;
  float* dinv = agg + (size_t)N * HID;
  _Float16* pu  = (_Float16*)(dinv + N);        // 64*128  halves
  _Float16* pf  = pu  + 64 * HID;               // 128*128 halves
  _Float16* pc1 = pf  + 128 * HID;
  _Float16* pc2 = pc1 + 128 * HID;

  float* exist  = (float*)d_out;
  float* rating = exist + P;

  const long long NH = (long long)N * HID;

  // 0) pack weights into WMMA B-fragment layout (f16)
  k_packB<<<32, 256, 0, stream>>>(u_W,  pu,  64);
  k_packB<<<64, 256, 0, stream>>>(f_W,  pf,  128);
  k_packB<<<64, 256, 0, stream>>>(c1_W, pc1, 128);
  k_packB<<<64, 256, 0, stream>>>(c2_W, pc2, 128);

  // 1) degrees -> dinv = rsqrt(in_deg + 1)
  k_zero<<<512, 256, 0, stream>>>(dinv, N);
  k_deg<<<2048, 256, 0, stream>>>(dst, E, dinv);
  k_dinv<<<(N + 255) / 256, 256, 0, stream>>>(dinv, N);

  // 2) input projections -> agg = x0
  k_gemm_wmma<true, 64><<<N_U / 16, 256, 0, stream>>>(u_feat, pu, u_b, agg, N_U);
  k_gemm_wmma<true, 128><<<N_F / 16, 256, 0, stream>>>(f_feat, pf, f_b,
                                                       agg + (size_t)N_U * HID, N_F);

  // 3) conv1: xw = x0 @ c1_W ; agg = segsum + self-loop + bias ; LN + leaky in place
  k_gemm_wmma<false, 128><<<N / 16, 256, 0, stream>>>(agg, pc1, nullptr, xw, N);
  k_zero<<<2048, 256, 0, stream>>>(agg, NH);
  k_aggregate<<<8192, 256, 0, stream>>>(xw, dinv, src, dst, agg, E);
  k_post<<<N / 8 + 1, 256, 0, stream>>>(agg, xw, dinv, c1_b, n1_g, n1_b, N);

  // 4) conv2 (same pattern) -> agg holds final embeddings x2
  k_gemm_wmma<false, 128><<<N / 16, 256, 0, stream>>>(agg, pc2, nullptr, xw, N);
  k_zero<<<2048, 256, 0, stream>>>(agg, NH);
  k_aggregate<<<8192, 256, 0, stream>>>(xw, dinv, src, dst, agg, E);
  k_post<<<N / 8 + 1, 256, 0, stream>>>(agg, xw, dinv, c2_b, n2_g, n2_b, N);

  // 5) edge scoring head
  k_predict<<<16384, 256, 0, stream>>>(agg, edge_u, edge_f, e_W, e_b, r_W, r_b,
                                       exist, rating, P);
}
